// FlashAttention_3805341024549
// MI455X (gfx1250) — matrix operand order repaired and checked by reading, not compiled, not run on hardware
//
#include <hip/hip_runtime.h>
#include <hip/hip_bf16.h>

typedef __attribute__((ext_vector_type(16))) __bf16 v16bf;
typedef __attribute__((ext_vector_type(8)))  __bf16 v8bf;
typedef __attribute__((ext_vector_type(8)))  float  v8f;

#define DIM    512
#define NHEAD  8
#define HDIM   64
#define BATCH  2
#define SEQ    4096
#define MTOT   (BATCH*SEQ)                      // 8192
#define NQKV   (3*DIM)                          // 1536
#define HSIZE  ((size_t)BATCH*NHEAD*SEQ*HDIM)   // 4194304 elems per Q/K/V tensor
#define KS     40   // padded LDS stride (elems) for 32-wide K tiles (16B aligned, bank-spread)
#define KVP    72   // padded LDS stride (elems) for 64-wide tiles

__device__ __forceinline__ __bf16 f2bf(float f) { return (__bf16)f; }

// LDS byte offset of a generic pointer to __shared__ memory: the LDS aperture
// places the LDS offset in addr[31:0] (ISA 10.2 aperture address calculation).
__device__ __forceinline__ unsigned lds_off(const void* p) {
  return (unsigned)(size_t)p;
}

// CDNA5 async DMA: 16B global -> LDS per lane, tracked by ASYNCcnt.
__device__ __forceinline__ void async_cp16(unsigned lds_byte, const __bf16* g) {
  unsigned long long ga = (unsigned long long)(size_t)g;
  asm volatile("global_load_async_to_lds_b128 %0, %1, off"
               :: "v"(lds_byte), "v"(ga) : "memory");
}
__device__ __forceinline__ void wait_async0() {
  asm volatile("s_wait_asynccnt 0x0" ::: "memory");
}

__device__ __forceinline__ void load16f(const float* __restrict__ g, float* t) {
  const float4* g4 = (const float4*)g;
  float4 f0 = g4[0], f1 = g4[1], f2 = g4[2], f3 = g4[3];
  t[0]=f0.x;  t[1]=f0.y;  t[2]=f0.z;  t[3]=f0.w;
  t[4]=f1.x;  t[5]=f1.y;  t[6]=f1.z;  t[7]=f1.w;
  t[8]=f2.x;  t[9]=f2.y;  t[10]=f2.z; t[11]=f2.w;
  t[12]=f3.x; t[13]=f3.y; t[14]=f3.z; t[15]=f3.w;
}

// A fragment (16x32 bf16, MxK): lane = (half<<4)|m ; K halves at 8*half and 16+8*half.
__device__ __forceinline__ v16bf frag_a(const __bf16* base, int stride, int lane) {
  const int m = lane & 15, hl = lane >> 4;
  const __bf16* p = base + m*stride + 8*hl;
  v8bf lo = *(const v8bf*)(p);
  v8bf hi = *(const v8bf*)(p + 16);
  v16bf r;
  #pragma unroll
  for (int i = 0; i < 8; ++i) { r[i] = lo[i]; r[i+8] = hi[i]; }
  return r;
}

// B fragment read from an n-major (B-transposed) tile: lane n holds K halves at 8*half and 16+8*half.
__device__ __forceinline__ v16bf frag_bt(const __bf16* base, int stride, int lane) {
  const int n = lane & 15, hl = lane >> 4;
  const __bf16* p = base + n*stride + 8*hl;
  v8bf lo = *(const v8bf*)(p);
  v8bf hi = *(const v8bf*)(p + 16);
  v16bf r;
  #pragma unroll
  for (int i = 0; i < 8; ++i) { r[i] = lo[i]; r[i+8] = hi[i]; }
  return r;
}

// ---------------------------------------------------------------------------
// Kernel 1: qkv = x @ w_qkv + b_qkv, scattered to bf16 Q/K/V in [B,H,N,D]
// ---------------------------------------------------------------------------
__global__ __launch_bounds__(256)
void qkv_gemm(const float* __restrict__ x, const float* __restrict__ w,
              const float* __restrict__ bias, __bf16* __restrict__ qkv) {
  __shared__ __align__(16) __bf16 As[128*KS];
  __shared__ __align__(16) __bf16 Bs[128*KS];
  const int tid = threadIdx.x, lane = tid & 31, wid = tid >> 5;
  const int wm = wid >> 1, wn = wid & 1;
  const int m0 = blockIdx.x * 128, n0 = blockIdx.y * 128;
  const int hl = lane >> 4, ln = lane & 15;

  v8f zero = {};
  v8f acc[2][4];
  #pragma unroll
  for (int i = 0; i < 2; ++i)
    #pragma unroll
    for (int j = 0; j < 4; ++j) acc[i][j] = zero;

  const int arow = tid >> 1, acol = (tid & 1) * 16;
  const int brow = tid >> 3, bcol = (tid & 7) * 16;

  for (int kb = 0; kb < DIM; kb += 32) {
    __syncthreads();
    if (kb + 32 < DIM) {   // hide next tile's latency behind barrier + 8 WMMAs
      __builtin_prefetch(x + (size_t)(m0 + arow)*DIM + (kb + 32) + acol, 0, 0);
      __builtin_prefetch(w + (size_t)(kb + 32 + brow)*NQKV + n0 + bcol, 0, 0);
    }
    {   // stage A (fp32 -> bf16)
      float t[16];
      load16f(x + (size_t)(m0 + arow)*DIM + kb + acol, t);
      v8bf lo, hi;
      #pragma unroll
      for (int i = 0; i < 8; ++i) { lo[i] = f2bf(t[i]); hi[i] = f2bf(t[8+i]); }
      *(v8bf*)(As + arow*KS + acol)     = lo;
      *(v8bf*)(As + arow*KS + acol + 8) = hi;
    }
    {   // stage B transposed (Bs[n][k]) with fp32 -> bf16
      float u[16];
      load16f(w + (size_t)(kb + brow)*NQKV + n0 + bcol, u);
      #pragma unroll
      for (int i = 0; i < 16; ++i) Bs[(bcol + i)*KS + brow] = f2bf(u[i]);
    }
    __syncthreads();

    v16bf af[2];
    #pragma unroll
    for (int mi = 0; mi < 2; ++mi)
      af[mi] = frag_a(As + (wm*32 + mi*16)*KS, KS, lane);
    #pragma unroll
    for (int ni = 0; ni < 4; ++ni) {
      v16bf bfr = frag_bt(Bs + (wn*64 + ni*16)*KS, KS, lane);
      #pragma unroll
      for (int mi = 0; mi < 2; ++mi)
        acc[mi][ni] = __builtin_amdgcn_wmma_f32_16x16x32_bf16(
            false, af[mi], false, bfr, (short)0, acc[mi][ni], false, false);
    }
  }

  // epilogue: add bias, scatter into head-major bf16 Q/K/V
  #pragma unroll
  for (int ni = 0; ni < 4; ++ni) {
    const int gc = n0 + wn*64 + ni*16 + ln;
    const float bv = bias[gc];
    const int which = gc >> 9, cc = gc & 511;
    const int hh = cc >> 6, dd = cc & 63;
    __bf16* dst = qkv + (size_t)which * HSIZE;
    #pragma unroll
    for (int mi = 0; mi < 2; ++mi)
      #pragma unroll
      for (int j = 0; j < 8; ++j) {
        const int gm = m0 + wm*32 + mi*16 + 8*hl + j;
        const int bb = gm >> 12, nn = gm & 4095;
        dst[(((size_t)(bb*NHEAD + hh))*SEQ + nn)*HDIM + dd] =
            f2bf(acc[mi][ni][j] + bv);
      }
  }
}

// ---------------------------------------------------------------------------
// Kernel 2: flash attention, one (b,h) x 128-query block per workgroup.
// K/V tiles double-buffered via GLOBAL_LOAD_ASYNC_TO_LDS (ASYNCcnt).
// ---------------------------------------------------------------------------
__global__ __launch_bounds__(256)
void attn_fwd(const __bf16* __restrict__ Qg, const __bf16* __restrict__ Kg,
              const __bf16* __restrict__ Vg, __bf16* __restrict__ Og) {
  __shared__ __align__(16) __bf16 Ksh[2][64*KVP];   // K tile  [key][d], ping-pong
  __shared__ __align__(16) __bf16 Vraw[2][64*64];   // V tile  [key][d], ping-pong (DMA dest)
  __shared__ __align__(16) __bf16 VTsh[64*KVP];     // V tile transposed [d][key]
  __shared__ __align__(16) __bf16 Psh[8*16*KVP];    // per-wave P scratch
  const int tid = threadIdx.x, lane = tid & 31, wid = tid >> 5;
  const int hl = lane >> 4, ln = lane & 15;
  const int q0 = blockIdx.x * 128;
  const int bh = blockIdx.y;
  const size_t base = (size_t)bh * SEQ * HDIM;

  const int sr = tid >> 2, sc = (tid & 3) * 16;     // staging: row 0..63, col {0,16,32,48}

  // Q fragments (16 rows per wave), pre-scaled by 1/sqrt(HDIM)=0.125 (exact in bf16)
  v16bf qf[2];
  {
    const __bf16* qrow = Qg + base + (size_t)(q0 + wid*16 + ln)*HDIM;
    #pragma unroll
    for (int c = 0; c < 2; ++c) {
      const __bf16* p = qrow + c*32 + 8*hl;
      v8bf lo = *(const v8bf*)p;
      v8bf hi = *(const v8bf*)(p + 16);
      v16bf r;
      #pragma unroll
      for (int i = 0; i < 8; ++i) {
        r[i]   = f2bf((float)lo[i] * 0.125f);
        r[i+8] = f2bf((float)hi[i] * 0.125f);
      }
      qf[c] = r;
    }
  }

  v8f zero = {};
  v8f o[4];
  #pragma unroll
  for (int dt = 0; dt < 4; ++dt) o[dt] = zero;
  float mrow[8], lrow[8];
  #pragma unroll
  for (int j = 0; j < 8; ++j) { mrow[j] = -1e30f; lrow[j] = 0.f; }

  __bf16* pw = Psh + wid*16*KVP;

  // async-stage one 64-key K and V tile into buffer `buf`
  auto stage = [&](int kbase, int buf) {
    const __bf16* gk = Kg + base + (size_t)(kbase + sr)*HDIM + sc;
    const unsigned dk = lds_off(&Ksh[buf][sr*KVP + sc]);
    async_cp16(dk,      gk);
    async_cp16(dk + 16, gk + 8);
    const __bf16* gv = Vg + base + (size_t)(kbase + sr)*HDIM + sc;
    const unsigned dv = lds_off(&Vraw[buf][sr*64 + sc]);
    async_cp16(dv,      gv);
    async_cp16(dv + 16, gv + 8);
  };

  stage(0, 0);   // prologue DMA

  for (int kb = 0; kb < SEQ; kb += 64) {
    const int cur = (kb >> 6) & 1;
    wait_async0();          // this wave's DMAs for tile `cur` have landed
    __syncthreads();        // everyone's have; VTsh free; prev buffer's readers done

    if (kb + 64 < SEQ) stage(kb + 64, cur ^ 1);   // overlap next DMA with compute

    {   // LDS->LDS transpose of V tile into VTsh[d][key]
      const __bf16* vr = &Vraw[cur][sr*64 + sc];
      v8bf v0 = *(const v8bf*)vr, v1 = *(const v8bf*)(vr + 8);
      #pragma unroll
      for (int i = 0; i < 8; ++i) VTsh[(sc + i)*KVP + sr]     = v0[i];
      #pragma unroll
      for (int i = 0; i < 8; ++i) VTsh[(sc + 8 + i)*KVP + sr] = v1[i];
    }

    // S = (Q*scale) @ K^T : 4 key-tiles of 16, K-dim = HDIM (2 frags of 32)
    v8f s[4];
    #pragma unroll
    for (int nt = 0; nt < 4; ++nt) {
      s[nt] = zero;
      #pragma unroll
      for (int c = 0; c < 2; ++c) {
        v16bf kf = frag_bt(&Ksh[cur][nt*16*KVP + c*32], KVP, lane);
        s[nt] = __builtin_amdgcn_wmma_f32_16x16x32_bf16(
            false, qf[c], false, kf, (short)0, s[nt], false, false);
      }
    }

    // online softmax (row state replicated across the 16 lanes sharing a row)
    #pragma unroll
    for (int j = 0; j < 8; ++j) {
      float mx = fmaxf(fmaxf(s[0][j], s[1][j]), fmaxf(s[2][j], s[3][j]));
      #pragma unroll
      for (int off = 8; off >= 1; off >>= 1)
        mx = fmaxf(mx, __shfl_xor(mx, off, 32));
      const float mnew  = fmaxf(mrow[j], mx);
      const float alpha = __expf(mrow[j] - mnew);
      float ps = 0.f;
      #pragma unroll
      for (int nt = 0; nt < 4; ++nt) {
        const float p = __expf(s[nt][j] - mnew);
        s[nt][j] = p;
        ps += p;
      }
      #pragma unroll
      for (int off = 8; off >= 1; off >>= 1)
        ps += __shfl_xor(ps, off, 32);
      lrow[j] = lrow[j]*alpha + ps;
      mrow[j] = mnew;
      #pragma unroll
      for (int dt = 0; dt < 4; ++dt) o[dt][j] *= alpha;
    }

    __syncthreads();        // VTsh fully written by all waves

    // C-layout -> A-layout transpose of P through per-wave LDS scratch
    #pragma unroll
    for (int j = 0; j < 8; ++j)
      #pragma unroll
      for (int nt = 0; nt < 4; ++nt)
        pw[(j + 8*hl)*KVP + nt*16 + ln] = f2bf(s[nt][j]);
    asm volatile("s_wait_dscnt 0x0" ::: "memory");   // same-wave LDS RAW

    v16bf pf[2];
    #pragma unroll
    for (int c = 0; c < 2; ++c) pf[c] = frag_a(pw + c*32, KVP, lane);
    #pragma unroll
    for (int dt = 0; dt < 4; ++dt)
      #pragma unroll
      for (int c = 0; c < 2; ++c) {
        v16bf vf = frag_bt(VTsh + dt*16*KVP + c*32, KVP, lane);
        o[dt] = __builtin_amdgcn_wmma_f32_16x16x32_bf16(
            false, pf[c], false, vf, (short)0, o[dt], false, false);
      }
  }

  // normalize + merge heads -> bf16 [B, N, C]
  const int bb = bh >> 3, hh = bh & 7;
  #pragma unroll
  for (int j = 0; j < 8; ++j) {
    const float inv = 1.f / lrow[j];
    const int row = q0 + wid*16 + 8*hl + j;
    #pragma unroll
    for (int dt = 0; dt < 4; ++dt) {
      const int col = hh*HDIM + dt*16 + ln;
      Og[((size_t)(bb*SEQ + row))*DIM + col] = f2bf(o[dt][j] * inv);
    }
  }
}

// ---------------------------------------------------------------------------
// Kernel 3: out = attn @ w_proj + b_proj  (A already bf16 -> async DMA; fp32 out)
// ---------------------------------------------------------------------------
__global__ __launch_bounds__(256)
void proj_gemm(const __bf16* __restrict__ a, const float* __restrict__ w,
               const float* __restrict__ bias, float* __restrict__ out) {
  __shared__ __align__(16) __bf16 As[128*KS];
  __shared__ __align__(16) __bf16 Bs[128*KS];
  const int tid = threadIdx.x, lane = tid & 31, wid = tid >> 5;
  const int wm = wid >> 1, wn = wid & 1;
  const int m0 = blockIdx.x * 128, n0 = blockIdx.y * 128;
  const int hl = lane >> 4, ln = lane & 15;

  v8f zero = {};
  v8f acc[2][4];
  #pragma unroll
  for (int i = 0; i < 2; ++i)
    #pragma unroll
    for (int j = 0; j < 4; ++j) acc[i][j] = zero;

  const int arow = tid >> 1, acol = (tid & 1) * 16;
  const int brow = tid >> 3, bcol = (tid & 7) * 16;

  for (int kb = 0; kb < DIM; kb += 32) {
    __syncthreads();
    if (kb + 32 < DIM)
      __builtin_prefetch(w + (size_t)(kb + 32 + brow)*DIM + n0 + bcol, 0, 0);
    {   // stage A via async DMA (bf16 straight copy, no VGPR round-trip)
      const __bf16* g = a + (size_t)(m0 + arow)*DIM + kb + acol;
      const unsigned da = lds_off(&As[arow*KS + acol]);
      async_cp16(da,      g);
      async_cp16(da + 16, g + 8);
    }
    {   // stage B transposed with fp32 -> bf16 (overlaps with the DMA above)
      float u[16];
      load16f(w + (size_t)(kb + brow)*DIM + n0 + bcol, u);
      #pragma unroll
      for (int i = 0; i < 16; ++i) Bs[(bcol + i)*KS + brow] = f2bf(u[i]);
    }
    wait_async0();
    __syncthreads();

    v16bf af[2];
    #pragma unroll
    for (int mi = 0; mi < 2; ++mi)
      af[mi] = frag_a(As + (wm*32 + mi*16)*KS, KS, lane);
    #pragma unroll
    for (int ni = 0; ni < 4; ++ni) {
      v16bf bfr = frag_bt(Bs + (wn*64 + ni*16)*KS, KS, lane);
      #pragma unroll
      for (int mi = 0; mi < 2; ++mi)
        acc[mi][ni] = __builtin_amdgcn_wmma_f32_16x16x32_bf16(
            false, af[mi], false, bfr, (short)0, acc[mi][ni], false, false);
    }
  }

  #pragma unroll
  for (int ni = 0; ni < 4; ++ni) {
    const int gc = n0 + wn*64 + ni*16 + ln;
    const float bv = bias[gc];
    #pragma unroll
    for (int mi = 0; mi < 2; ++mi)
      #pragma unroll
      for (int j = 0; j < 8; ++j) {
        const int gm = m0 + wm*32 + mi*16 + 8*hl + j;
        out[(size_t)gm*DIM + gc] = acc[mi][ni][j] + bv;
      }
  }
}

// ---------------------------------------------------------------------------
extern "C" void kernel_launch(void* const* d_in, const int* in_sizes, int n_in,
                              void* d_out, int out_size, void* d_ws, size_t ws_size,
                              hipStream_t stream) {
  const float* x      = (const float*)d_in[0];
  const float* w_qkv  = (const float*)d_in[1];
  const float* b_qkv  = (const float*)d_in[2];
  const float* w_proj = (const float*)d_in[3];
  const float* b_proj = (const float*)d_in[4];
  float* out = (float*)d_out;

  // workspace layout (bf16): Q | K | V | attn_out  => 4 * 8 MiB = 32 MiB
  __bf16* qkv  = (__bf16*)d_ws;
  __bf16* attn = qkv + 3 * HSIZE;

  qkv_gemm<<<dim3(MTOT/128, NQKV/128), 256, 0, stream>>>(x, w_qkv, b_qkv, qkv);
  attn_fwd<<<dim3(SEQ/128, BATCH*NHEAD), 256, 0, stream>>>(
      qkv, qkv + HSIZE, qkv + 2*HSIZE, attn);
  proj_gemm<<<dim3(MTOT/128, DIM/128), 256, 0, stream>>>(attn, w_proj, b_proj, out);

  (void)in_sizes; (void)n_in; (void)out_size; (void)ws_size;
}
